// CorrBase_86457691668630
// MI455X (gfx1250) — matrix operand order repaired and checked by reading, not compiled, not run on hardware
//
#include <hip/hip_runtime.h>
#include <hip/hip_bf16.h>

// Problem constants (match reference)
#define BB 8
#define CC 256
#define HH 96
#define WW 128
#define RR 4
#define HP (HH + 2 * RR)   // 104 padded rows
#define WP 144             // 128 + 2*4 pad + 8 slack so B-tile1 (x0+31) never faults
#define ND 81              // (2r+1)^2 displacements

typedef __attribute__((ext_vector_type(16))) _Float16 v16h;
typedef __attribute__((ext_vector_type(8)))  _Float16 v8h;
typedef __attribute__((ext_vector_type(8)))  float    v8f;

static constexpr size_t FRN_HALVES = (size_t)BB * HH * WW * CC;  // 25,165,824
static constexpr size_t FQP_HALVES = (size_t)BB * HP * WP * CC;  // 30,670,848
static constexpr size_t FRN_BYTES  = FRN_HALVES * 2;             // 50,331,648 (256B aligned)
static constexpr size_t FQP_BYTES  = FQP_HALVES * 2;             // 61,341,696

// Assemble a 16-half fragment from two contiguous-16B chunks at p and p+off.
__device__ __forceinline__ v16h ld_frag(const _Float16* p, int off_halves) {
  v8h lo = *(const v8h*)p;
  v8h up = *(const v8h*)(p + off_halves);
  return __builtin_shufflevector(lo, up, 0, 1, 2, 3, 4, 5, 6, 7, 8, 9, 10, 11,
                                 12, 13, 14, 15);
}

// ---------------------------------------------------------------------------
// Normalize (B,C,H,W) f32 -> pixel-major (B,Ho,Wo,C) f16, optionally padded.
// One block per (b,y) row; thread x owns pixel x. Pass 1: coalesced sum of
// squares over channels. Pass 2 re-reads (row stays hot in L2), scales,
// converts to f16, stores 16B per thread.
// ---------------------------------------------------------------------------
__global__ __launch_bounds__(128) void corr_nrm_kernel(
    const float* __restrict__ in, _Float16* __restrict__ outp,
    int Ho, int Wo, int pad) {
  const int b = blockIdx.x / HH;
  const int y = blockIdx.x - b * HH;
  const int x = threadIdx.x;
  const float* src = in + (size_t)b * CC * HH * WW + (size_t)y * WW + x;

  float ss = 0.0f;
#pragma unroll 8
  for (int c = 0; c < CC; ++c) {
    float v = src[(size_t)c * HH * WW];
    ss = fmaf(v, v, ss);
  }
  const float inv = 1.0f / fmaxf(sqrtf(ss), 1e-12f);  // matches F.normalize eps

  _Float16* dst = outp + ((size_t)(b * Ho + y + pad) * Wo + (x + pad)) * CC;
#pragma unroll 2
  for (int c0 = 0; c0 < CC; c0 += 8) {
    v8h h;
#pragma unroll
    for (int j = 0; j < 8; ++j)
      h[j] = (_Float16)(src[(size_t)(c0 + j) * HH * WW] * inv);
    *(v8h*)(dst + c0) = h;
  }
}

// ---------------------------------------------------------------------------
// WMMA correlation. One wave per 16-pixel strip (block = 8 waves = full row).
// A tile: 16 pixels x 256 ch of frn. Per dy: two 16-col B tiles of padded fq
// row y+dy; diagonals of the two 16x16 D tiles give all 9 dx displacements.
// B fragments are software double-buffered so the next K-step's loads are in
// flight while the current pair of WMMAs executes.
// ---------------------------------------------------------------------------
__global__ __launch_bounds__(256) void corr_wmma_kernel(
    const _Float16* __restrict__ A, const _Float16* __restrict__ Q,
    float* __restrict__ out) {
  __shared__ float st[8][144];  // per-wave staging: [dx 0..8][m 0..15]

  const int b    = blockIdx.x / HH;
  const int y    = blockIdx.x - b * HH;
  const int wv   = threadIdx.x >> 5;
  const int lane = threadIdx.x & 31;
  const int x0   = wv * 16;
  const int n    = lane & 15;   // A: row m ; B: col n ; C/D: col n
  const int hi   = lane >> 4;

  // --- preload all 8 A fragments (K = 256) -------------------------------
  // f16 A 16x32 layout: lane L<16 holds row L, K{0..7,16..23}; lane L+16
  // holds K{8..15,24..31}  ->  two b128 loads at +hi*8 and +16+hi*8 halves.
  const _Float16* aBase =
      A + ((size_t)(b * HH + y) * WW + x0 + n) * CC + hi * 8;
  v16h afrag[8];
#pragma unroll
  for (int kc = 0; kc < 8; ++kc) afrag[kc] = ld_frag(aBase + kc * 32, 16);

  const float scale = 1.0f / (float)CC;  // jnp.mean over channels
  // f16 B 32x16 layout: lane L holds col L&15, K{0..7,16..23} (L<16) or
  // K{8..15,24..31} (L>=16) -> two b128 loads at +hi*8 and +16+hi*8 halves.
  const _Float16* qRow =
      Q + ((size_t)(b * HP + y) * WP + x0 + n) * CC + hi * 8;

  for (int dy = 0; dy < 9; ++dy) {
    const _Float16* q0 = qRow + (size_t)dy * WP * CC;   // tile 0: cols x0..x0+15
    const _Float16* q1 = q0 + (size_t)16 * CC;          // tile 1: cols x0+16..x0+31
    if (dy < 8) __builtin_prefetch(q0 + (size_t)WP * CC, 0, 1);

    v8f c0 = {0.f, 0.f, 0.f, 0.f, 0.f, 0.f, 0.f, 0.f};
    v8f c1 = {0.f, 0.f, 0.f, 0.f, 0.f, 0.f, 0.f, 0.f};

    // Double-buffered K loop: loads for kc+1 issue before WMMAs of kc.
    v16h bf0 = ld_frag(q0, 16);
    v16h bf1 = ld_frag(q1, 16);
#pragma unroll
    for (int kc = 0; kc < 8; ++kc) {
      v16h nb0, nb1;
      if (kc < 7) {
        nb0 = ld_frag(q0 + (kc + 1) * 32, 16);
        nb1 = ld_frag(q1 + (kc + 1) * 32, 16);
      }
      c0 = __builtin_amdgcn_wmma_f32_16x16x32_f16(false, afrag[kc], false, bf0,
                                                  (short)0, c0, false, false);
      c1 = __builtin_amdgcn_wmma_f32_16x16x32_f16(false, afrag[kc], false, bf1,
                                                  (short)0, c1, false, false);
      if (kc < 7) {
        bf0 = nb0;
        bf1 = nb1;
      }
    }

    // --- scatter needed diagonals into LDS -------------------------------
    // C/D layout: lane holds col n; VGPR v holds row m = v + 8*(lane>>4).
#pragma unroll
    for (int v = 0; v < 8; ++v) {
      const int m  = v + hi * 8;
      const int d0 = n - m;        // tile 0 diagonal -> dx index
      if (d0 >= 0 && d0 <= 8) st[wv][d0 * 16 + m] = c0[v] * scale;
      const int d1 = n + 16 - m;   // tile 1 diagonal -> dx index
      if (d1 >= 0 && d1 <= 8) st[wv][d1 * 16 + m] = c1[v] * scale;
    }

    // --- coalesced writeout: out[b, dy*9+dx, y, x0+m] --------------------
    float* obase = out + (((size_t)b * ND + dy * 9) * HH + y) * WW + x0;
    for (int t = lane; t < 144; t += 32) {
      const int dx = t >> 4;
      const int m  = t & 15;
      obase[(size_t)dx * HH * WW + m] = st[wv][t];
    }
  }
}

extern "C" void kernel_launch(void* const* d_in, const int* in_sizes, int n_in,
                              void* d_out, int out_size, void* d_ws,
                              size_t ws_size, hipStream_t stream) {
  const float* fr = (const float*)d_in[0];
  const float* fq = (const float*)d_in[1];
  float* out = (float*)d_out;

  _Float16* frn = (_Float16*)d_ws;
  _Float16* fqp = (_Float16*)((char*)d_ws + FRN_BYTES);

  // Zero the padded fq buffer (borders + right slack must be exactly 0).
  hipMemsetAsync(fqp, 0, FQP_BYTES, stream);

  // Normalize + cast to f16, pixel-major.
  corr_nrm_kernel<<<BB * HH, 128, 0, stream>>>(fr, frn, HH, WW, 0);
  corr_nrm_kernel<<<BB * HH, 128, 0, stream>>>(fq, fqp, HP, WP, RR);

  // WMMA correlation: one block per (b,y) row, one wave per 16-pixel strip.
  corr_wmma_kernel<<<BB * HH, 256, 0, stream>>>(frn, fqp, out);
}
